// MultiHeadAttention_64647847739885
// MI455X (gfx1250) — matrix operand order repaired and checked by reading, not compiled, not run on hardware
//
#include <hip/hip_runtime.h>
#include <hip/hip_bf16.h>
#include <math.h>

// ---------------------------------------------------------------------------
// MI455X (gfx1250) multi-head attention pipeline, all matmuls on
// v_wmma_f32_16x16x32_bf16 (K=32 per issue, f32 accumulate).
// Roofline: ~52 GFLOP over ~70MB live data; 192MB L2 holds the activation
// working set; the 256MB score tensor round-trips HBM in ~22us @ 23.3TB/s,
// cheaper than recomputing scores for the query-axis softmax.
// New this round: global_load_async_to_lds_b128 staging (ASYNCcnt path) for
// the bf16-A GEMM, vectorized tile staging, global_prefetch of next K tile.
// Workspace requirement: ~297 MB (dominated by the f32 score tensor).
// ---------------------------------------------------------------------------

typedef __bf16 bf16_t;
typedef __bf16 v16bf __attribute__((ext_vector_type(16)));
typedef __bf16 v8bf  __attribute__((ext_vector_type(8)));
typedef float  v8f   __attribute__((ext_vector_type(8)));

#define B_      4
#define T_      1024
#define DM_     1024
#define H_      16
#define DK_     64
#define BT_     (B_ * T_)

__device__ __forceinline__ v16bf cat8(v8bf lo, v8bf hi) {
  return __builtin_shufflevector(lo, hi, 0,1,2,3,4,5,6,7,8,9,10,11,12,13,14,15);
}

__device__ __forceinline__ v8f wmma_bf16(v16bf a, v16bf b, v8f c) {
  // (neg_a, A, neg_b, B, c_mod, C, reuse_a, reuse_b)
  return __builtin_amdgcn_wmma_f32_16x16x32_bf16(false, a, false, b, (short)0, c,
                                                 false, false);
}

// ---------------------------------------------------------------------------
// Kernel 1: masked column-softmax of Wq/Wk over the DMODEL axis.
// Input  W[h][d][k] (f32); output cat-layout Wc[d][h*64+k] (bf16) so all 16
// heads fuse into one 1024x1024 GEMM B-matrix. Valid entries: d >= k.
// One block per (q/k, h, k) column; 256-thread LDS tree reduction.
// ---------------------------------------------------------------------------
__global__ void __launch_bounds__(256)
wsoftmax_kernel(const float* __restrict__ Wq, const float* __restrict__ Wk,
                const int* __restrict__ maskout,
                bf16_t* __restrict__ Oq, bf16_t* __restrict__ Ok) {
  __shared__ float red[256];
  const int col = blockIdx.x;                 // 0 .. 2*H*DK-1
  const int qk  = (col >= H_ * DK_) ? 1 : 0;
  const int c   = col - qk * H_ * DK_;
  const int h   = c / DK_, k = c % DK_;
  const float* src = (qk ? Wk : Wq) + (size_t)h * DM_ * DK_ + k;  // stride DK_ over d
  bf16_t* dst = (qk ? Ok : Oq) + h * DK_ + k;                     // stride DM_ over d
  const int tid = threadIdx.x;

  if (*maskout == 0) {
    for (int d = tid; d < DM_; d += 256)
      dst[(size_t)d * DM_] = (bf16_t)src[(size_t)d * DK_];
    return;
  }
  // pass 1: max over valid d (d >= k)
  float m = -INFINITY;
  for (int d = k + tid; d < DM_; d += 256)
    m = fmaxf(m, src[(size_t)d * DK_]);
  red[tid] = m; __syncthreads();
  for (int s = 128; s > 0; s >>= 1) {
    if (tid < s) red[tid] = fmaxf(red[tid], red[tid + s]);
    __syncthreads();
  }
  m = red[0]; __syncthreads();
  // pass 2: sum of exp
  float z = 0.f;
  for (int d = k + tid; d < DM_; d += 256)
    z += __expf(src[(size_t)d * DK_] - m);
  red[tid] = z; __syncthreads();
  for (int s = 128; s > 0; s >>= 1) {
    if (tid < s) red[tid] += red[tid + s];
    __syncthreads();
  }
  const float rz = 1.0f / red[0];
  // pass 3: write normalized probs (0 above the triangle)
  for (int d = tid; d < DM_; d += 256) {
    float v = (d >= k) ? __expf(src[(size_t)d * DK_] - m) * rz : 0.f;
    dst[(size_t)d * DM_] = (bf16_t)v;
  }
}

// ---------------------------------------------------------------------------
// Kernel 2: Wv -> cat bf16 layout, Wo -> bf16. 1M threads do both.
// ---------------------------------------------------------------------------
__global__ void __launch_bounds__(256)
wconvert_kernel(const float* __restrict__ Wv, const float* __restrict__ Wo,
                bf16_t* __restrict__ Ov, bf16_t* __restrict__ Oo) {
  const int idx = blockIdx.x * 256 + threadIdx.x;  // 0 .. DM_*DM_-1
  const int d = idx >> 10, c = idx & 1023;
  const int h = c >> 6, v = c & 63;
  Ov[idx] = (bf16_t)Wv[(size_t)h * DM_ * DK_ + (size_t)d * DK_ + v];
  Oo[idx] = (bf16_t)Wo[idx];
}

// ---------------------------------------------------------------------------
// Kernel 3/7: tiled bf16 WMMA GEMM, C(4096x1024) = A(4096x1024) * B(1024x1024),
// all strides 1024. Block = 256 threads (8 waves), tile 128x128, K step 32.
// Wave layout 4(M)x2(N): each wave owns 32x64 = 2x4 WMMA accumulators.
// B staged transposed in LDS so B fragments are contiguous 16B ds loads.
// A_F32: f32 activations, converted to bf16 through VGPRs (float4 loads).
// !A_F32: A tile DMA'd global->LDS with global_load_async_to_lds_b128
//         (ASYNCcnt), no VGPR round-trip. C_F32: f32 store (d_out).
// ---------------------------------------------------------------------------
template <bool A_F32, bool C_F32>
__global__ void __launch_bounds__(256)
gemm_kernel(const void* __restrict__ Ap, const bf16_t* __restrict__ Bp,
            void* __restrict__ Cp) {
  constexpr int LD = 1024;
  __shared__ __align__(16) bf16_t As[128 * 40];   // [row][k], stride 40
  __shared__ __align__(16) bf16_t Bs[128 * 40];   // transposed: [n][k], stride 40
  const int tid  = threadIdx.x;
  const int lane = tid & 31, wid = tid >> 5;
  const int half = lane >> 4, l16 = lane & 15;
  const int wm0  = (wid & 3) * 32;
  const int wn0  = (wid >> 2) * 64;
  const int bm0  = blockIdx.y * 128;
  const int bn0  = blockIdx.x * 128;

  v8f acc[2][4];
#pragma unroll
  for (int mi = 0; mi < 2; ++mi)
#pragma unroll
    for (int ni = 0; ni < 4; ++ni)
#pragma unroll
      for (int i = 0; i < 8; ++i) acc[mi][ni][i] = 0.f;

  for (int kt = 0; kt < 1024; kt += 32) {
    if (A_F32) {
      // stage A tile 128x32 f32 -> bf16, vectorized float4 loads
      const float* Af = (const float*)Ap;
#pragma unroll
      for (int it = 0; it < 4; ++it) {
        const int slot = tid + it * 256;          // 1024 float4 slots
        const int r = slot >> 3, c4 = (slot & 7) * 4;
        const float4 x = *(const float4*)&Af[(size_t)(bm0 + r) * LD + kt + c4];
        bf16_t* dst = &As[r * 40 + c4];
        dst[0] = (bf16_t)x.x; dst[1] = (bf16_t)x.y;
        dst[2] = (bf16_t)x.z; dst[3] = (bf16_t)x.w;
      }
    } else {
      // async DMA A tile (128 rows x 64B) straight into LDS; 512 16B chunks
      const bf16_t* Ab = (const bf16_t*)Ap;
#pragma unroll
      for (int it = 0; it < 2; ++it) {
        const int chunk = tid + it * 256;
        const int r = chunk >> 2, e = (chunk & 3) * 8;   // 4 chunks per row
        const unsigned lds = (unsigned)(size_t)&As[r * 40 + e];
        const void* gp = &Ab[(size_t)(bm0 + r) * LD + kt + e];
        asm volatile("global_load_async_to_lds_b128 %0, %1, off"
                     :: "v"(lds), "v"(gp) : "memory");
      }
    }
    // stage B tile 32x128 transposed; 16B vector loads, scalar LDS stores
#pragma unroll
    for (int it = 0; it < 2; ++it) {
      const int slot = tid + it * 256;            // 512 8-elem slots
      const int kk = slot >> 4, n0 = (slot & 15) * 8;
      const v8bf x = *(const v8bf*)&Bp[(size_t)(kt + kk) * LD + bn0 + n0];
#pragma unroll
      for (int i = 0; i < 8; ++i) Bs[(n0 + i) * 40 + kk] = x[i];
    }
    // prefetch next B K-tile while this one computes
    if (kt + 32 < 1024)
      __builtin_prefetch(&Bp[(size_t)(kt + 32 + (tid >> 3)) * LD + bn0], 0, 0);
    if (!A_F32)
      asm volatile("s_wait_asynccnt 0x0" ::: "memory");
    __syncthreads();

    v16bf afr[2];
#pragma unroll
    for (int mi = 0; mi < 2; ++mi) {
      const bf16_t* ap = &As[(wm0 + mi * 16 + l16) * 40 + 8 * half];
      v8bf lo = *(const v8bf*)ap;
      v8bf hi = *(const v8bf*)(ap + 16);   // K = 16 + 8*half ..
      afr[mi] = cat8(lo, hi);
    }
#pragma unroll
    for (int ni = 0; ni < 4; ++ni) {
      const bf16_t* bp = &Bs[(wn0 + ni * 16 + l16) * 40 + 8 * half];
      v8bf lo = *(const v8bf*)bp;
      v8bf hi = *(const v8bf*)(bp + 16);
      const v16bf bfr = cat8(lo, hi);
#pragma unroll
      for (int mi = 0; mi < 2; ++mi)
        acc[mi][ni] = wmma_bf16(afr[mi], bfr, acc[mi][ni]);
    }
    __syncthreads();
  }
  // store: D element i lives at (row = 8*half + i, col = l16) of each tile
#pragma unroll
  for (int mi = 0; mi < 2; ++mi)
#pragma unroll
    for (int ni = 0; ni < 4; ++ni) {
      const int col = bn0 + wn0 + ni * 16 + l16;
#pragma unroll
      for (int i = 0; i < 8; ++i) {
        const int row = bm0 + wm0 + mi * 16 + 8 * half + i;
        if (C_F32) ((float*)Cp)[(size_t)row * LD + col] = acc[mi][ni][i];
        else       ((bf16_t*)Cp)[(size_t)row * LD + col] = (bf16_t)acc[mi][ni][i];
      }
    }
}

// ---------------------------------------------------------------------------
// Kernel 4: scores[bh][t][s] = (1/8) * sum_k Qp[bt][h*64+k]*Kp[bs][h*64+k].
// Block: 8 waves, each wave a 16(t)x64(s) strip; K=64 -> 2 WMMA steps.
// Fragments loaded directly from global (contiguous 16B per lane; hot in L2).
// ---------------------------------------------------------------------------
__global__ void __launch_bounds__(256)
scores_kernel(const bf16_t* __restrict__ Qp, const bf16_t* __restrict__ Kp,
              float* __restrict__ S) {
  const int tid = threadIdx.x, lane = tid & 31, wid = tid >> 5;
  const int half = lane >> 4, l16 = lane & 15;
  const int bh = blockIdx.z, b = bh >> 4, h = bh & 15;
  const int t0 = blockIdx.y * 128 + wid * 16;
  const int s0 = blockIdx.x * 64;

  v8f acc[4];
#pragma unroll
  for (int ni = 0; ni < 4; ++ni)
#pragma unroll
    for (int i = 0; i < 8; ++i) acc[ni][i] = 0.f;

  const int row = t0 + l16;
  const bf16_t* qrow = Qp + ((size_t)(b * T_ + row)) * DM_ + h * 64;
#pragma unroll
  for (int ko = 0; ko < 64; ko += 32) {
    v8bf alo = *(const v8bf*)(qrow + ko + 8 * half);
    v8bf ahi = *(const v8bf*)(qrow + ko + 16 + 8 * half);
    const v16bf afr = cat8(alo, ahi);
#pragma unroll
    for (int ni = 0; ni < 4; ++ni) {
      const int sc = s0 + ni * 16 + l16;
      const bf16_t* krow = Kp + ((size_t)(b * T_ + sc)) * DM_ + h * 64 + ko + 8 * half;
      v8bf blo = *(const v8bf*)krow;
      v8bf bhi = *(const v8bf*)(krow + 16);
      acc[ni] = wmma_bf16(afr, cat8(blo, bhi), acc[ni]);
    }
  }
  float* Sbh = S + (size_t)bh * T_ * T_;
#pragma unroll
  for (int ni = 0; ni < 4; ++ni) {
    const int col = s0 + ni * 16 + l16;
#pragma unroll
    for (int i = 0; i < 8; ++i) {
      const int r = t0 + 8 * half + i;
      Sbh[(size_t)r * T_ + col] = acc[ni][i] * 0.125f;  // 1/sqrt(64)
    }
  }
}

// ---------------------------------------------------------------------------
// Kernel 5: per-column (key axis s) streaming logsumexp over t (query axis).
// One thread per (bh, s); adjacent threads read adjacent addresses.
// ---------------------------------------------------------------------------
__global__ void __launch_bounds__(256)
colstats_kernel(const float* __restrict__ S, float* __restrict__ cm,
                float* __restrict__ crz) {
  const int idx = blockIdx.x * 256 + threadIdx.x;  // bh*T + s
  const int bh = idx >> 10, s = idx & 1023;
  const float* col = S + (size_t)bh * T_ * T_ + s;
  float m = -INFINITY, z = 0.f;
  for (int t = 0; t < T_; ++t) {
    const float x = col[(size_t)t * T_];
    if (x > m) { z = z * __expf(m - x) + 1.f; m = x; }
    else       { z += __expf(x - m); }
  }
  cm[idx]  = m;
  crz[idx] = 1.0f / z;
}

// ---------------------------------------------------------------------------
// Kernel 6: ctx[bt][h*64+v] = sum_s exp(S - m_s)*rz_s * Vp[bs][h*64+v].
// Probabilities formed in-register (f32 -> bf16) directly in A-fragment
// layout; V tile (32s x 64v) staged transposed in LDS each K step.
// ---------------------------------------------------------------------------
__global__ void __launch_bounds__(256)
ctx_kernel(const float* __restrict__ S, const float* __restrict__ cm,
           const float* __restrict__ crz, const bf16_t* __restrict__ Vp,
           bf16_t* __restrict__ CTX) {
  __shared__ __align__(16) bf16_t VsT[64 * 40];  // [v][s_local], stride 40
  const int tid = threadIdx.x, lane = tid & 31, wid = tid >> 5;
  const int half = lane >> 4, l16 = lane & 15;
  const int bh = blockIdx.z, b = bh >> 4, h = bh & 15;
  const int t0 = blockIdx.y * 128 + wid * 16;

  v8f acc[4];
#pragma unroll
  for (int ni = 0; ni < 4; ++ni)
#pragma unroll
    for (int i = 0; i < 8; ++i) acc[ni][i] = 0.f;

  const int row = t0 + l16;
  const float* Srow = S + (size_t)bh * T_ * T_ + (size_t)row * T_;
  const float* m_  = cm  + bh * T_;
  const float* rz_ = crz + bh * T_;

  for (int s0 = 0; s0 < T_; s0 += 32) {
    // stage V tile 32(s) x 64(v) transposed into LDS
    {
      int idx = tid;
#pragma unroll
      for (int it = 0; it < 8; ++it, idx += 256) {
        const int sl = idx >> 6, v = idx & 63;
        VsT[v * 40 + sl] =
            Vp[((size_t)(b * T_ + s0 + sl)) * DM_ + h * 64 + v];
      }
    }
    __syncthreads();
    // A fragment: softmax probs for row t over s = s0..s0+31 (lane layout)
    v16bf afr;
    {
      const int sA = s0 + 8 * half;
#pragma unroll
      for (int i = 0; i < 8; ++i)
        afr[i] = (bf16_t)(__expf(Srow[sA + i] - m_[sA + i]) * rz_[sA + i]);
      const int sB = s0 + 16 + 8 * half;
#pragma unroll
      for (int i = 0; i < 8; ++i)
        afr[8 + i] = (bf16_t)(__expf(Srow[sB + i] - m_[sB + i]) * rz_[sB + i]);
    }
#pragma unroll
    for (int ni = 0; ni < 4; ++ni) {
      const bf16_t* bp = &VsT[(ni * 16 + l16) * 40 + 8 * half];
      v8bf lo = *(const v8bf*)bp;
      v8bf hi = *(const v8bf*)(bp + 16);
      acc[ni] = wmma_bf16(afr, cat8(lo, hi), acc[ni]);
    }
    __syncthreads();
  }
#pragma unroll
  for (int ni = 0; ni < 4; ++ni) {
    const int v = ni * 16 + l16;
#pragma unroll
    for (int i = 0; i < 8; ++i) {
      const int r = t0 + 8 * half + i;
      CTX[((size_t)(b * T_ + r)) * DM_ + h * 64 + v] = (bf16_t)acc[ni][i];
    }
  }
}

// ---------------------------------------------------------------------------
// Launch: inputs = [Q, K, V, Wq, Wk, Wv, Wo, maskout]; output f32 (B,T,DMODEL).
// Workspace layout (256B aligned blocks), total ~297 MB.
// ---------------------------------------------------------------------------
extern "C" void kernel_launch(void* const* d_in, const int* in_sizes, int n_in,
                              void* d_out, int out_size, void* d_ws,
                              size_t ws_size, hipStream_t stream) {
  (void)in_sizes; (void)n_in; (void)out_size; (void)ws_size;
  const float* Q  = (const float*)d_in[0];
  const float* K  = (const float*)d_in[1];
  const float* V  = (const float*)d_in[2];
  const float* Wq = (const float*)d_in[3];
  const float* Wk = (const float*)d_in[4];
  const float* Wv = (const float*)d_in[5];
  const float* Wo = (const float*)d_in[6];
  const int* maskout = (const int*)d_in[7];
  float* out = (float*)d_out;

  char* ws = (char*)d_ws;
  size_t off = 0;
  auto take = [&](size_t bytes) -> char* {
    char* p = ws + off;
    off += (bytes + 255) & ~(size_t)255;
    return p;
  };
  const size_t wbytes = (size_t)DM_ * DM_ * sizeof(bf16_t);   // 2 MB
  const size_t abytes = (size_t)BT_ * DM_ * sizeof(bf16_t);   // 8 MB
  bf16_t* Wqc = (bf16_t*)take(wbytes);
  bf16_t* Wkc = (bf16_t*)take(wbytes);
  bf16_t* Wvc = (bf16_t*)take(wbytes);
  bf16_t* Woc = (bf16_t*)take(wbytes);
  bf16_t* Qp  = (bf16_t*)take(abytes);
  bf16_t* Kp  = (bf16_t*)take(abytes);
  bf16_t* Vp  = (bf16_t*)take(abytes);
  bf16_t* CTX = (bf16_t*)take(abytes);
  float*  Ssc = (float*)take((size_t)B_ * H_ * T_ * T_ * sizeof(float)); // 256 MB
  float*  cmx = (float*)take((size_t)B_ * H_ * T_ * sizeof(float));
  float*  crz = (float*)take((size_t)B_ * H_ * T_ * sizeof(float));

  // 1) weight softmax (tril-masked over DMODEL) + layout to head-concat bf16
  wsoftmax_kernel<<<2 * H_ * DK_, 256, 0, stream>>>(Wq, Wk, maskout, Wqc, Wkc);
  // 2) Wv / Wo conversion
  wconvert_kernel<<<(DM_ * DM_) / 256, 256, 0, stream>>>(Wv, Wo, Wvc, Woc);
  // 3) projections: (4096x1024 f32) x (1024x1024 bf16) -> bf16
  dim3 gg(DM_ / 128, BT_ / 128);
  gemm_kernel<true, false><<<gg, 256, 0, stream>>>((const void*)Q, Wqc, (void*)Qp);
  gemm_kernel<true, false><<<gg, 256, 0, stream>>>((const void*)K, Wkc, (void*)Kp);
  gemm_kernel<true, false><<<gg, 256, 0, stream>>>((const void*)V, Wvc, (void*)Vp);
  // 4) scores (scaled by 1/sqrt(dk)) -> f32 workspace
  scores_kernel<<<dim3(T_ / 64, T_ / 128, B_ * H_), 256, 0, stream>>>(Qp, Kp, Ssc);
  // 5) per-key-column logsumexp over the query axis
  colstats_kernel<<<(B_ * H_ * T_) / 256, 256, 0, stream>>>(Ssc, cmx, crz);
  // 6) ctx = softmax(S, axis=t) @ V  -> bf16 head-concat layout
  ctx_kernel<<<dim3(1, T_ / 128, B_ * H_), 256, 0, stream>>>(Ssc, cmx, crz, Vp, CTX);
  // 7) output projection: (4096x1024 bf16) x (1024x1024 bf16) -> f32 d_out
  gemm_kernel<false, true><<<gg, 256, 0, stream>>>((const void*)CTX, Woc, (void*)out);
}
